// TokenFlowModel_3238405341740
// MI455X (gfx1250) — matrix operand order repaired and checked by reading, not compiled, not run on hardware
//
#include <hip/hip_runtime.h>
#include <cstdint>
#include <cstddef>

// ---------------------------------------------------------------------------
// Types
// ---------------------------------------------------------------------------
typedef __bf16 bf16;
typedef __attribute__((ext_vector_type(16))) __bf16 v16bf;
typedef __attribute__((ext_vector_type(8)))  float  v8f;
typedef __attribute__((ext_vector_type(4)))  unsigned int u32x4;

union FragU { u32x4 u[2]; v16bf v; };

// Model dims (compile-time, match reference)
constexpr int S_   = 2048;
constexpr int D_   = 2048;
constexpr int H_   = 16;
constexpr int KVH_ = 8;
constexpr int HD_  = 128;
constexpr int FH_  = 5632;
constexpr int MOD3 = 3 * D_;          // 6144
constexpr int NEXP = 256;             // N_ expansion (S / 8)
constexpr int KVD  = KVH_ * HD_;      // 1024

// ---------------------------------------------------------------------------
// WMMA / async helpers
// ---------------------------------------------------------------------------
__device__ __forceinline__ v8f zero8() {
  v8f z = {0.f, 0.f, 0.f, 0.f, 0.f, 0.f, 0.f, 0.f};
  return z;
}

__device__ __forceinline__ v8f wmma_bf16(v16bf a, v16bf b, v8f c) {
  return __builtin_amdgcn_wmma_f32_16x16x32_bf16(false, a, false, b, (short)0, c,
                                                 false, false);
}

// A-fragment (16x32 bf16): lane l holds row (l&15); K = {kb..kb+7} U {kb+16..kb+23},
// kb = (l>>4)*8. Two 16B loads.
__device__ __forceinline__ v16bf frag_a_lds(const bf16* row, int hi) {
  FragU t;
  const int kb = hi * 8;
  t.u[0] = *(const u32x4*)(row + kb);
  t.u[1] = *(const u32x4*)(row + kb + 16);
  return t.v;
}

// B-fragment (32x16 bf16) from k-contiguous tile Bs[n][k]: lane l holds col
// (l&15); K = {kb..kb+7} U {kb+16..kb+23}, kb = (l>>4)*8. Two 16B loads.
__device__ __forceinline__ v16bf frag_b_lds(const bf16* row, int hi) {
  FragU t;
  const int kb = hi * 8;
  t.u[0] = *(const u32x4*)(row + kb);
  t.u[1] = *(const u32x4*)(row + kb + 16);
  return t.v;
}

// Async DMA: 16 bytes per lane, global -> LDS, tracked by ASYNCcnt.
__device__ __forceinline__ void async_copy_b128(uint32_t lds_addr, const void* gptr) {
  asm volatile("global_load_async_to_lds_b128 %0, %1, off"
               :: "v"(lds_addr), "v"((unsigned long long)(uintptr_t)gptr)
               : "memory");
}
__device__ __forceinline__ void wait_async0() {
  asm volatile("s_wait_asynccnt 0x0" ::: "memory");
}

// Low 32 bits of a flat LDS address are the LDS byte offset (aperture rule).
template <typename T>
__device__ __forceinline__ uint32_t lds_off(T* p) {
  return (uint32_t)(uintptr_t)p;
}

// ---------------------------------------------------------------------------
// Pack / small kernels
// ---------------------------------------------------------------------------

// Transposed bf16 weight pack: wt[n*K + k] = (bf16)w[k*N + n]
__global__ __launch_bounds__(256) void pack_wt_kernel(
    const float* __restrict__ w, bf16* __restrict__ wt, int K, int N) {
  const size_t idx = (size_t)blockIdx.x * 256 + threadIdx.x;  // over N*K
  const int n = (int)(idx / K);
  const int k = (int)(idx % K);
  wt[idx] = (bf16)w[(size_t)k * N + n];
}

// mod = silu(vec) @ mod_w + mod_b   (8 x 6144 output)
__global__ __launch_bounds__(256) void mod_gemm_kernel(
    const float* __restrict__ vec, const float* __restrict__ mw,
    const float* __restrict__ mb, float* __restrict__ mod) {
  const int idx = blockIdx.x * 256 + threadIdx.x;  // 0..49151
  const int n = idx % MOD3;
  const int m = idx / MOD3;
  const float* vr = vec + (size_t)m * D_;
  float acc = 0.f;
  for (int d = 0; d < D_; ++d) {
    float s = vr[d];
    s = s / (1.f + __expf(-s));
    acc += s * mw[(size_t)d * MOD3 + n];
  }
  mod[idx] = acc + mb[n];
}

// RMSNorm (+ optional shift/scale modulation), bf16 output. One block per row.
__global__ __launch_bounds__(256) void rmsnorm_kernel(
    const float* __restrict__ x, const float* __restrict__ w,
    const float* __restrict__ shift, const float* __restrict__ scale,
    bf16* __restrict__ out) {
  const int row = blockIdx.x;
  const int tid = threadIdx.x;
  const float* xr = x + (size_t)row * D_;
  float ss = 0.f;
  for (int i = tid; i < D_; i += 256) { float v = xr[i]; ss += v * v; }
  #pragma unroll
  for (int m = 16; m >= 1; m >>= 1) ss += __shfl_xor(ss, m, 32);
  __shared__ float red[8];
  if ((tid & 31) == 0) red[tid >> 5] = ss;
  __syncthreads();
  float tot = red[0];
  #pragma unroll
  for (int i = 1; i < 8; ++i) tot += red[i];
  const float r = rsqrtf(tot * (1.f / (float)D_) + 1e-6f);
  const int g = row / NEXP;
  bf16* orow = out + (size_t)row * D_;
  if (shift != nullptr) {
    const float* sh = shift + (size_t)g * MOD3;
    const float* sc = scale + (size_t)g * MOD3;
    for (int i = tid; i < D_; i += 256) {
      const float v = xr[i] * r * w[i];
      orow[i] = (bf16)((1.f + sc[i]) * v + sh[i]);
    }
  } else {
    for (int i = tid; i < D_; i += 256) orow[i] = (bf16)(xr[i] * r * w[i]);
  }
}

// RoPE + bf16 pack: dst[...] = rope(src[...])
__global__ __launch_bounds__(256) void rope_pack_kernel(
    const float* __restrict__ src, bf16* __restrict__ dst,
    const float* __restrict__ cosb, const float* __restrict__ sinb, int nheads) {
  const int idx = blockIdx.x * 256 + threadIdx.x;
  const int total = S_ * nheads * (HD_ / 2);
  if (idx >= total) return;
  const int i = idx & 63;
  const int h = (idx >> 6) % nheads;
  const int s = idx / (64 * nheads);
  const size_t base = ((size_t)s * nheads + h) * HD_ + 2 * i;
  const float c = cosb[s * 64 + i];
  const float sn = sinb[s * 64 + i];
  const float xr = src[base], xi = src[base + 1];
  dst[base]     = (bf16)(xr * c - xi * sn);
  dst[base + 1] = (bf16)(xr * sn + xi * c);
}

// V transpose-pack: vtb[c*S + s] = (bf16)v[s*KVD + c]
__global__ __launch_bounds__(256) void pack_vt_kernel(
    const float* __restrict__ v, bf16* __restrict__ vtb) {
  const int idx = blockIdx.x * 256 + threadIdx.x;  // over S*KVD
  const int s = idx >> 10;
  const int c = idx & (KVD - 1);
  vtb[(size_t)c * S_ + s] = (bf16)v[idx];
}

// ff = silu(t1) * t3 -> bf16
__global__ __launch_bounds__(256) void silu_mul_kernel(
    const float* __restrict__ t1, const float* __restrict__ t3,
    bf16* __restrict__ ff, int n) {
  const int idx = blockIdx.x * 256 + threadIdx.x;
  if (idx < n) {
    const float a = t1[idx];
    ff[idx] = (bf16)((a / (1.f + __expf(-a))) * t3[idx]);
  }
}

// ---------------------------------------------------------------------------
// Tiled bf16-WMMA GEMM with async-to-LDS double buffering.
// C[MxN](f32) = A[MxK](bf16) @ Bt[NxK](bf16, pre-transposed)
// mode 0: C = acc ; mode 1: C = res + gate[row/256][col]*acc ; mode 2: C = res + acc
// Block: 128x128 tile, 256 threads (8 waves, 2x4), K-step 32.
// ---------------------------------------------------------------------------
__global__ __launch_bounds__(256) void gemm_kernel(
    const bf16* __restrict__ A, const bf16* __restrict__ Bt,
    float* __restrict__ C, int M, int N, int K, int mode,
    const float* __restrict__ res, const float* __restrict__ gate) {
  __shared__ __align__(16) bf16 As[2][128][40];  // [buf][m][k], padded stride
  __shared__ __align__(16) bf16 Bs[2][128][40];  // [buf][n][k], padded stride
  const int tid = threadIdx.x;
  const int lane = tid & 31, wid = tid >> 5;
  const int wm = wid & 1, wn = wid >> 1;  // 2 x 4 wave grid
  const int bm = blockIdx.y * 128, bn = blockIdx.x * 128;
  const int ln = lane & 15, hi = lane >> 4;
  const uint32_t asBase = lds_off(&As[0][0][0]);
  const uint32_t bsBase = lds_off(&Bs[0][0][0]);
  constexpr uint32_t BUFB = 128 * 40 * 2;  // bytes per buffer

  // DMA one 128x32 bf16 tile of A and Bt into LDS buffer `buf`.
  // 512 16B-chunks per matrix; 2 per thread.
  auto issue_tile = [&](int kt, int buf) {
    #pragma unroll
    for (int i = 0; i < 2; ++i) {
      const int chunk = tid + 256 * i;          // 0..511
      const int r = chunk >> 2;                 // tile row
      const int e = (chunk & 3) * 8;            // bf16 element offset in row
      async_copy_b128(asBase + buf * BUFB + r * 80 + e * 2,
                      A + (size_t)(bm + r) * K + kt * 32 + e);
      async_copy_b128(bsBase + buf * BUFB + r * 80 + e * 2,
                      Bt + (size_t)(bn + r) * K + kt * 32 + e);
    }
  };

  v8f acc[4][2];
  #pragma unroll
  for (int i = 0; i < 4; ++i)
    #pragma unroll
    for (int j = 0; j < 2; ++j) acc[i][j] = zero8();

  const int nkt = K >> 5;
  issue_tile(0, 0);
  wait_async0();
  __syncthreads();

  for (int kt = 0; kt < nkt; ++kt) {
    const int cur = kt & 1;
    if (kt + 1 < nkt) issue_tile(kt + 1, 1 - cur);   // DMA overlaps compute

    v16bf af[4], bfv[2];
    #pragma unroll
    for (int mi = 0; mi < 4; ++mi)
      af[mi] = frag_a_lds(&As[cur][wm * 64 + mi * 16 + ln][0], hi);
    #pragma unroll
    for (int ni = 0; ni < 2; ++ni)
      bfv[ni] = frag_b_lds(&Bs[cur][wn * 32 + ni * 16 + ln][0], hi);
    #pragma unroll
    for (int mi = 0; mi < 4; ++mi)
      #pragma unroll
      for (int ni = 0; ni < 2; ++ni)
        acc[mi][ni] = wmma_bf16(af[mi], bfv[ni], acc[mi][ni]);

    wait_async0();
    __syncthreads();
  }

  // Epilogue. C fragment: VGPR j -> row j (lanes 0-15) / row j+8 (lanes 16-31)
  #pragma unroll
  for (int mi = 0; mi < 4; ++mi)
    #pragma unroll
    for (int ni = 0; ni < 2; ++ni)
      #pragma unroll
      for (int j = 0; j < 8; ++j) {
        const int row = bm + wm * 64 + mi * 16 + j + 8 * hi;
        const int col = bn + wn * 32 + ni * 16 + ln;
        const size_t idx = (size_t)row * N + col;
        float vv = acc[mi][ni][j];
        if (mode == 1)      vv = res[idx] + gate[(size_t)(row / NEXP) * MOD3 + col] * vv;
        else if (mode == 2) vv = res[idx] + vv;
        C[idx] = vv;
      }
}

// ---------------------------------------------------------------------------
// Flash attention (causal, GQA 2:1), bf16 inputs, bf16 output.
// q: (S, H*HD) bf16 ; k: (S, KVD) bf16 ; vt: (KVD, S) bf16 transposed.
// Grid: (S/64, H). Block: 128 threads (4 waves x 16 query rows).
// ---------------------------------------------------------------------------
__global__ __launch_bounds__(128) void attn_kernel(
    const bf16* __restrict__ q, const bf16* __restrict__ k,
    const bf16* __restrict__ vt, bf16* __restrict__ o) {
  __shared__ __align__(16) bf16 Ks[32][136];     // [keypos][hd]
  __shared__ __align__(16) bf16 Vt[128][40];     // [hd][keypos]
  __shared__ __align__(16) bf16 Ps[4][16][40];   // per-wave P scratch
  const int tid = threadIdx.x;
  const int lane = tid & 31, wid = tid >> 5;
  const int ln = lane & 15, hi = lane >> 4;
  const int head = blockIdx.y;
  const int kvh = head >> 1;                      // n_rep = 2
  const int qblk = blockIdx.x;                    // 64 query rows per block
  const int qrow0 = qblk * 64 + wid * 16;
  const float scl = 0.088388347648318447f;        // 1/sqrt(128)
  const uint32_t ksBase = lds_off(&Ks[0][0]);
  const uint32_t vtBase = lds_off(&Vt[0][0]);

  // Load Q fragments once (A layout), direct bf16 16B loads.
  v16bf qf[4];
  {
    const bf16* qbase = q + (size_t)(qrow0 + ln) * (H_ * HD_) + head * HD_;
    #pragma unroll
    for (int c = 0; c < 4; ++c) {
      FragU t;
      t.u[0] = *(const u32x4*)(qbase + c * 32 + hi * 8);
      t.u[1] = *(const u32x4*)(qbase + c * 32 + hi * 8 + 16);
      qf[c] = t.v;
    }
  }

  v8f oacc[8];
  #pragma unroll
  for (int n = 0; n < 8; ++n) oacc[n] = zero8();
  float mrow[8], lrow[8];
  #pragma unroll
  for (int j = 0; j < 8; ++j) { mrow[j] = -1e30f; lrow[j] = 0.f; }

  const int ntiles = qblk * 2 + 2;  // causal bound over 32-key tiles
  for (int kt = 0; kt < ntiles; ++kt) {
    const int kp0 = kt * 32;
    // Async DMA K tile (32 keys x 128 hd) and V^T tile (128 hd x 32 keys).
    // Each tile = 512 16B chunks; 4 per thread per tile.
    #pragma unroll
    for (int i = 0; i < 4; ++i) {
      const int chunk = tid + 128 * i;            // 0..511
      const int kp = chunk >> 4;                  // K: 16 chunks per key row
      const int e8 = (chunk & 15) * 8;            // element offset in row
      async_copy_b128(ksBase + kp * 272 + e8 * 2,
                      k + (size_t)(kp0 + kp) * KVD + kvh * HD_ + e8);
      const int hd = chunk >> 2;                  // V: 4 chunks per hd row
      const int c8 = (chunk & 3) * 8;
      async_copy_b128(vtBase + hd * 80 + c8 * 2,
                      vt + (size_t)(kvh * HD_ + hd) * S_ + kp0 + c8);
    }
    wait_async0();
    __syncthreads();

    // scores S = Q @ K^T  (two 16-col fragments covering 32 keys)
    v8f s0 = zero8(), s1 = zero8();
    #pragma unroll
    for (int c = 0; c < 4; ++c) {
      const v16bf b0 = frag_b_lds(&Ks[ln][c * 32], hi);
      const v16bf b1 = frag_b_lds(&Ks[16 + ln][c * 32], hi);
      s0 = wmma_bf16(qf[c], b0, s0);
      s1 = wmma_bf16(qf[c], b1, s1);
    }

    // online softmax per row
    #pragma unroll
    for (int j = 0; j < 8; ++j) {
      const int qi = qrow0 + j + 8 * hi;
      float a0 = s0[j] * scl + ((kp0 + ln)      > qi ? -1e9f : 0.f);
      float a1 = s1[j] * scl + ((kp0 + 16 + ln) > qi ? -1e9f : 0.f);
      float mj = fmaxf(a0, a1);
      mj = fmaxf(mj, __shfl_xor(mj, 1, 32));
      mj = fmaxf(mj, __shfl_xor(mj, 2, 32));
      mj = fmaxf(mj, __shfl_xor(mj, 4, 32));
      mj = fmaxf(mj, __shfl_xor(mj, 8, 32));
      const float mnew = fmaxf(mrow[j], mj);
      const float alpha = __expf(mrow[j] - mnew);
      a0 = __expf(a0 - mnew);
      a1 = __expf(a1 - mnew);
      float rs = a0 + a1;
      rs += __shfl_xor(rs, 1, 32);
      rs += __shfl_xor(rs, 2, 32);
      rs += __shfl_xor(rs, 4, 32);
      rs += __shfl_xor(rs, 8, 32);
      lrow[j] = lrow[j] * alpha + rs;
      mrow[j] = mnew;
      #pragma unroll
      for (int n = 0; n < 8; ++n) oacc[n][j] *= alpha;
      // stash P (C layout -> LDS) for A-layout reload
      const int m = j + 8 * hi;
      Ps[wid][m][ln]      = (bf16)a0;
      Ps[wid][m][16 + ln] = (bf16)a1;
    }

    // O += P @ V (K=32)
    const v16bf pf = frag_a_lds(&Ps[wid][ln][0], hi);
    #pragma unroll
    for (int n = 0; n < 8; ++n) {
      const v16bf vb = frag_b_lds(&Vt[n * 16 + ln][0], hi);
      oacc[n] = wmma_bf16(pf, vb, oacc[n]);
    }
    __syncthreads();
  }

  // normalize + store (bf16)
  #pragma unroll
  for (int n = 0; n < 8; ++n)
    #pragma unroll
    for (int j = 0; j < 8; ++j) {
      const int row = qrow0 + j + 8 * hi;
      o[(size_t)row * (H_ * HD_) + head * HD_ + n * 16 + ln] =
          (bf16)(oacc[n][j] / lrow[j]);
    }
}

// ---------------------------------------------------------------------------
// Host launcher
// ---------------------------------------------------------------------------
extern "C" void kernel_launch(void* const* d_in, const int* in_sizes, int n_in,
                              void* d_out, int out_size, void* d_ws, size_t ws_size,
                              hipStream_t stream) {
  (void)in_sizes; (void)n_in; (void)out_size; (void)ws_size;
  const float* x    = (const float*)d_in[0];
  const float* vec  = (const float*)d_in[1];
  const float* cosb = (const float*)d_in[2];
  const float* sinb = (const float*)d_in[3];
  // d_in[4] = mask (computed analytically in the attention kernel)
  const float* wq   = (const float*)d_in[5];
  const float* wk   = (const float*)d_in[6];
  const float* wv   = (const float*)d_in[7];
  const float* wo   = (const float*)d_in[8];
  const float* w1   = (const float*)d_in[9];
  const float* w2   = (const float*)d_in[10];
  const float* w3   = (const float*)d_in[11];
  const float* mw   = (const float*)d_in[12];
  const float* mb   = (const float*)d_in[13];
  const float* anw  = (const float*)d_in[14];
  const float* fnw  = (const float*)d_in[15];
  float* out = (float*)d_out;
  float* ws = (float*)d_ws;

  // Workspace layout (in float slots; bf16 buffers use 2 elements per slot)
  constexpr size_t SD    = (size_t)S_ * D_;          // 4,194,304
  constexpr size_t SKV   = (size_t)S_ * KVD;         // 2,097,152
  constexpr size_t SF    = (size_t)S_ * FH_;         // 11,534,336
  constexpr size_t OFF_MOD = 0;
  constexpr size_t OFF_HBF = OFF_MOD + (size_t)8 * MOD3;   // bf16 S*D -> SD/2 slots
  constexpr size_t OFF_WQT = OFF_HBF + SD / 2;             // bf16 D*D
  constexpr size_t OFF_WKT = OFF_WQT + SD / 2;             // bf16 KVD*D
  constexpr size_t OFF_WVT = OFF_WKT + SKV / 2;
  constexpr size_t OFF_WOT = OFF_WVT + SKV / 2;            // bf16 D*D
  constexpr size_t OFF_W1T = OFF_WOT + SD / 2;             // bf16 FH*D
  constexpr size_t OFF_W3T = OFF_W1T + SF / 2;
  constexpr size_t OFF_W2T = OFF_W3T + SF / 2;             // bf16 D*FH
  constexpr size_t OFF_QB  = OFF_W2T + SF / 2;             // f32 S*D
  constexpr size_t OFF_KB  = OFF_QB + SD;                  // f32 S*KVD
  constexpr size_t OFF_VB  = OFF_KB + SKV;                 // f32 S*KVD
  constexpr size_t OFF_QBF = OFF_VB + SKV;                 // bf16 S*D
  constexpr size_t OFF_KBF = OFF_QBF + SD / 2;             // bf16 S*KVD
  constexpr size_t OFF_VTB = OFF_KBF + SKV / 2;            // bf16 KVD*S
  constexpr size_t OFF_OBF = OFF_VTB + SKV / 2;            // bf16 S*D
  constexpr size_t OFF_X2  = OFF_OBF + SD / 2;             // f32 S*D
  constexpr size_t OFF_T1  = OFF_X2 + SD;                  // f32 S*FH
  constexpr size_t OFF_T3  = OFF_T1 + SF;                  // f32 S*FH

  float* mod = ws + OFF_MOD;
  bf16* hbf  = (bf16*)(ws + OFF_HBF);
  bf16* wqT  = (bf16*)(ws + OFF_WQT);
  bf16* wkT  = (bf16*)(ws + OFF_WKT);
  bf16* wvT  = (bf16*)(ws + OFF_WVT);
  bf16* woT  = (bf16*)(ws + OFF_WOT);
  bf16* w1T  = (bf16*)(ws + OFF_W1T);
  bf16* w3T  = (bf16*)(ws + OFF_W3T);
  bf16* w2T  = (bf16*)(ws + OFF_W2T);
  float* qb  = ws + OFF_QB;
  float* kb  = ws + OFF_KB;
  float* vb  = ws + OFF_VB;
  bf16* qbf  = (bf16*)(ws + OFF_QBF);
  bf16* kbf  = (bf16*)(ws + OFF_KBF);
  bf16* vtb  = (bf16*)(ws + OFF_VTB);
  bf16* obf  = (bf16*)(ws + OFF_OBF);
  float* x2  = ws + OFF_X2;
  float* t1  = ws + OFF_T1;
  float* t3  = ws + OFF_T3;
  bf16* ffbf = (bf16*)(ws + OFF_QB);  // alias q/k/v f32 region (free after attn)

  // 0. weight packs (bf16, transposed to [N][K])
  pack_wt_kernel<<<(int)(SD / 256), 256, 0, stream>>>(wq, wqT, D_, D_);
  pack_wt_kernel<<<(int)(SKV / 256), 256, 0, stream>>>(wk, wkT, D_, KVD);
  pack_wt_kernel<<<(int)(SKV / 256), 256, 0, stream>>>(wv, wvT, D_, KVD);
  pack_wt_kernel<<<(int)(SD / 256), 256, 0, stream>>>(wo, woT, D_, D_);
  pack_wt_kernel<<<(int)(SF / 256), 256, 0, stream>>>(w1, w1T, D_, FH_);
  pack_wt_kernel<<<(int)(SF / 256), 256, 0, stream>>>(w3, w3T, D_, FH_);
  pack_wt_kernel<<<(int)(SF / 256), 256, 0, stream>>>(w2, w2T, FH_, D_);
  // 1. modulation
  mod_gemm_kernel<<<(8 * MOD3) / 256, 256, 0, stream>>>(vec, mw, mb, mod);
  // 2. h = (1+scale)*rmsnorm(x)*attn_norm_w + shift  (bf16)
  rmsnorm_kernel<<<S_, 256, 0, stream>>>(x, anw, mod, mod + D_, hbf);
  // 3-5. QKV projections (f32 out for RoPE)
  gemm_kernel<<<dim3(D_ / 128, S_ / 128), 256, 0, stream>>>(hbf, wqT, qb, S_, D_, D_, 0, nullptr, nullptr);
  gemm_kernel<<<dim3(KVD / 128, S_ / 128), 256, 0, stream>>>(hbf, wkT, kb, S_, KVD, D_, 0, nullptr, nullptr);
  gemm_kernel<<<dim3(KVD / 128, S_ / 128), 256, 0, stream>>>(hbf, wvT, vb, S_, KVD, D_, 0, nullptr, nullptr);
  // 6. RoPE + bf16 pack; V transpose-pack
  rope_pack_kernel<<<(S_ * H_ * 64) / 256, 256, 0, stream>>>(qb, qbf, cosb, sinb, H_);
  rope_pack_kernel<<<(S_ * KVH_ * 64) / 256, 256, 0, stream>>>(kb, kbf, cosb, sinb, KVH_);
  pack_vt_kernel<<<(int)(SKV / 256), 256, 0, stream>>>(vb, vtb);
  // 7. flash attention (bf16 in/out)
  attn_kernel<<<dim3(S_ / 64, H_), 128, 0, stream>>>(qbf, kbf, vtb, obf);
  // 8. x2 = x + gate * (o @ wo)
  gemm_kernel<<<dim3(D_ / 128, S_ / 128), 256, 0, stream>>>(obf, woT, x2, S_, D_, D_, 1, x, mod + 2 * D_);
  // 9. hn = rmsnorm(x2)*ffn_norm_w (bf16)
  rmsnorm_kernel<<<S_, 256, 0, stream>>>(x2, fnw, nullptr, nullptr, hbf);
  // 10-11. FFN up projections
  gemm_kernel<<<dim3(FH_ / 128, S_ / 128), 256, 0, stream>>>(hbf, w1T, t1, S_, FH_, D_, 0, nullptr, nullptr);
  gemm_kernel<<<dim3(FH_ / 128, S_ / 128), 256, 0, stream>>>(hbf, w3T, t3, S_, FH_, D_, 0, nullptr, nullptr);
  // 12. gated activation (bf16)
  silu_mul_kernel<<<(int)(SF / 256), 256, 0, stream>>>(t1, t3, ffbf, (int)SF);
  // 13. out = x2 + ff @ w2
  gemm_kernel<<<dim3(D_ / 128, S_ / 128), 256, 0, stream>>>(ffbf, w2T, out, S_, D_, FH_, 2, x2, nullptr);
}
